// MHA_KNN_V_58849641890550
// MI455X (gfx1250) — matrix operand order repaired and checked by reading, not compiled, not run on hardware
//
#include <hip/hip_runtime.h>
#include <hip/hip_bf16.h>
#include <cstdint>

#define BATCH 4
#define NPTS  8192
#define EMB   256
#define HEADS 8
#define KNN   6
#define DH    32
#define TILE  16   // points per workgroup in MHA kernel

typedef __attribute__((ext_vector_type(16))) __bf16 v16bf;
typedef __attribute__((ext_vector_type(8)))  float  v8f;

union ABvec { v16bf v; uint4 u[2]; };

// A-matrix (16x32 bf16) per CDNA5 ISA layout:
// lanes 0-15: row m=lane, K = kbase+0..7 (VGPR0-3), kbase+16..23 (VGPR4-7)
// lanes 16-31: row m=lane-16, K = kbase+8..15, kbase+24..31
__device__ __forceinline__ v16bf load_A(const __hip_bfloat16* base, int row_stride,
                                        int kbase, int lane) {
  const int m  = lane & 15;
  const int hi = lane >> 4;
  const __hip_bfloat16* row = base + (size_t)m * row_stride;
  ABvec r;
  r.u[0] = *reinterpret_cast<const uint4*>(row + kbase + hi * 8);
  r.u[1] = *reinterpret_cast<const uint4*>(row + kbase + 16 + hi * 8);
  return r.v;
}

// B-matrix (32x16 bf16): lane n=lane&15 holds column n; lanes 0-15 K=kbase+0..7, kbase+16..23,
// lanes 16-31 K=kbase+8..15, kbase+24..31.  Column n of B == row n of the (transposed) weight.
__device__ __forceinline__ v16bf load_B(const __hip_bfloat16* base, int row_stride,
                                        int kbase, int lane) {
  const int n  = lane & 15;
  const int hi = lane >> 4;
  const __hip_bfloat16* row = base + (size_t)n * row_stride;
  ABvec r;
  r.u[0] = *reinterpret_cast<const uint4*>(row + kbase + hi * 8);
  r.u[1] = *reinterpret_cast<const uint4*>(row + kbase + 16 + hi * 8);
  return r.v;
}

__device__ __forceinline__ v8f wmma_bf16(v8f acc, v16bf a, v16bf b) {
  // D = A*B + C, f32 accumulate
  return __builtin_amdgcn_wmma_f32_16x16x32_bf16(false, a, false, b,
                                                 (short)0, acc, false, false);
}

// ---------------------------------------------------------------------------
// Kernel 1: convert weights to bf16; Wk is stored transposed so all B-operands
// of every WMMA read contiguous rows.
// ---------------------------------------------------------------------------
__global__ __launch_bounds__(256)
void prep_weights(const float* __restrict__ in_proj, const float* __restrict__ out_proj,
                  __hip_bfloat16* __restrict__ Wq, __hip_bfloat16* __restrict__ WkT,
                  __hip_bfloat16* __restrict__ Wv, __hip_bfloat16* __restrict__ Wo) {
  const int i = blockIdx.x * 256 + threadIdx.x;       // 0 .. 65535
  const int f = i >> 8, e = i & 255;
  Wq[i]            = __float2bfloat16(in_proj[i]);
  WkT[e * EMB + f] = __float2bfloat16(in_proj[(EMB + f) * EMB + e]);
  Wv[i]            = __float2bfloat16(in_proj[2 * EMB * EMB + i]);
  Wo[i]            = __float2bfloat16(out_proj[i]);
}

__global__ __launch_bounds__(256)
void zero_kernel(float* __restrict__ p, int n) {
  const int i = blockIdx.x * 256 + threadIdx.x;
  if (i < n) p[i] = 0.0f;
}

// ---------------------------------------------------------------------------
// Kernel 2: brute-force KNN (squared L2, ascending, self first).
// One thread = one query point; candidates stream through a 256-point LDS tile.
// ---------------------------------------------------------------------------
__global__ __launch_bounds__(256)
void knn_kernel(const float* __restrict__ xv, int* __restrict__ idx_out) {
  __shared__ float cs[256][4];                        // x, y, z, |p|^2
  const int b = blockIdx.x / (NPTS / 256);
  const int q = (blockIdx.x % (NPTS / 256)) * 256 + threadIdx.x;
  const float* xvb = xv + (size_t)b * NPTS * 3;

  const float qx = xvb[q * 3 + 0], qy = xvb[q * 3 + 1], qz = xvb[q * 3 + 2];
  const float qsq = qx * qx + qy * qy + qz * qz;

  float bd[KNN];
  int   bi[KNN];
#pragma unroll
  for (int k = 0; k < KNN; ++k) { bd[k] = 3.4e38f; bi[k] = 0; }

  for (int t0 = 0; t0 < NPTS; t0 += 256) {
    __syncthreads();
    {
      const int c = t0 + threadIdx.x;
      const float cx = xvb[c * 3 + 0], cy = xvb[c * 3 + 1], cz = xvb[c * 3 + 2];
      cs[threadIdx.x][0] = cx; cs[threadIdx.x][1] = cy; cs[threadIdx.x][2] = cz;
      cs[threadIdx.x][3] = cx * cx + cy * cy + cz * cz;
    }
    __syncthreads();
    for (int c = 0; c < 256; ++c) {
      const float d = qsq + cs[c][3]
                    - 2.0f * (qx * cs[c][0] + qy * cs[c][1] + qz * cs[c][2]);
      if (d < bd[KNN - 1]) {
        int p = KNN - 1;
        while (p > 0 && bd[p - 1] > d) { bd[p] = bd[p - 1]; bi[p] = bi[p - 1]; --p; }
        bd[p] = d; bi[p] = t0 + c;
      }
    }
  }
#pragma unroll
  for (int k = 0; k < KNN; ++k)
    idx_out[((size_t)b * NPTS + q) * KNN + k] = bi[k];
}

// ---------------------------------------------------------------------------
// Kernel 3: fused MHA over 16-point tiles, bf16 WMMA with f32 accumulate.
// ---------------------------------------------------------------------------
__global__ __launch_bounds__(256)
void mha_kernel(const float* __restrict__ x, const int* __restrict__ idx,
                const __hip_bfloat16* __restrict__ Wq, const __hip_bfloat16* __restrict__ WkT,
                const __hip_bfloat16* __restrict__ Wv, const __hip_bfloat16* __restrict__ Wo,
                float* __restrict__ xout, float* __restrict__ sums, float* __restrict__ counts) {
  extern __shared__ char smem_raw[];
  __hip_bfloat16* x_bf  = (__hip_bfloat16*)smem_raw;             // [16][256]      8 KB
  __hip_bfloat16* k_bf  = x_bf + TILE * EMB;                     // [16][6][256]  48 KB
  __hip_bfloat16* q_bf  = k_bf + TILE * KNN * EMB;               // [16][256]      8 KB (also out_bf)
  __hip_bfloat16* big   = q_bf + TILE * EMB;                     // [16][8][256]  64 KB (qk then vbar)
  float* out_f  = (float*)(big + TILE * HEADS * EMB);            // [16][256]     16 KB
  float* attn_s = out_f + TILE * EMB;                            // [16][8][6]     3 KB
  float* wavg   = attn_s + TILE * HEADS * KNN;                   // [16][6]
  int*   idx_s  = (int*)(wavg + TILE * KNN);                     // [16][6]

  const int tid  = threadIdx.x;
  const int lane = tid & 31;
  const int wave = tid >> 5;                                     // 0..7
  const int b    = blockIdx.x / (NPTS / TILE);
  const int v0   = (blockIdx.x % (NPTS / TILE)) * TILE;
  const float* xb = x + (size_t)b * NPTS * EMB;

  // ---- Stage 0: load x tile, idx, gather neighbor rows (bf16) ----
  for (int i = tid; i < TILE * EMB; i += 256)
    x_bf[i] = __float2bfloat16(xb[(size_t)(v0 + (i >> 8)) * EMB + (i & 255)]);
  if (tid < TILE * KNN)
    idx_s[tid] = idx[((size_t)b * NPTS + v0 + tid / KNN) * KNN + tid % KNN];
  __syncthreads();
  for (int i = tid; i < TILE * KNN * EMB; i += 256) {
    const int row = i >> 8;                                      // m*KNN + k
    k_bf[i] = __float2bfloat16(xb[(size_t)idx_s[row] * EMB + (i & 255)]);
  }
  __syncthreads();

  // ---- Stage 1: qp = x @ Wq^T   (16 N-tiles, 2 per wave) ----
#pragma unroll
  for (int j = 0; j < 2; ++j) {
    const int nt = wave * 2 + j;
    v8f acc = {};
#pragma unroll
    for (int kk = 0; kk < 8; ++kk) {
      v16bf a  = load_A(x_bf, EMB, kk * 32, lane);
      v16bf bm = load_B(Wq + (size_t)nt * 16 * EMB, EMB, kk * 32, lane);
      acc = wmma_bf16(acc, a, bm);
    }
    const int n  = nt * 16 + (lane & 15);
    const int mb = (lane >> 4) * 8;
#pragma unroll
    for (int r = 0; r < 8; ++r)
      q_bf[(mb + r) * EMB + n] = __float2bfloat16(acc[r]);
  }
  __syncthreads();

  // ---- Stage 2: qk[m,h,:] = qp_h[16,32] @ Wk_h[32,256]  (single K-chunk WMMA)
  //      128 jobs (8 heads x 16 N-tiles), 16 per wave ----
#pragma unroll
  for (int j = 0; j < 16; ++j) {
    const int job = wave * 16 + j;
    const int h = job >> 4, nt = job & 15;
    v16bf a  = load_A(q_bf, EMB, h * 32, lane);
    v16bf bm = load_B(WkT + (size_t)nt * 16 * EMB, EMB, h * 32, lane);
    v8f acc = {};
    acc = wmma_bf16(acc, a, bm);
    const int n  = nt * 16 + (lane & 15);
    const int mb = (lane >> 4) * 8;
#pragma unroll
    for (int r = 0; r < 8; ++r)
      big[((mb + r) * HEADS + h) * EMB + n] = __float2bfloat16(acc[r]);
  }
  __syncthreads();

  // ---- Stage 3: scores = qk . k_raw / sqrt(Dh); softmax over K; head-avg ----
  for (int t = tid; t < TILE * HEADS * KNN; t += 256) {
    const int m = t / (HEADS * KNN);
    const int h = (t / KNN) % HEADS;
    const int k = t % KNN;
    const __hip_bfloat16* qk = &big[(m * HEADS + h) * EMB];
    const __hip_bfloat16* kr = &k_bf[(m * KNN + k) * EMB];
    float s = 0.0f;
#pragma unroll 8
    for (int e = 0; e < EMB; ++e)
      s += __bfloat162float(qk[e]) * __bfloat162float(kr[e]);
    attn_s[t] = s * 0.17677669529663687f;                        // 1/sqrt(32)
  }
  __syncthreads();
  if (tid < TILE * HEADS) {
    float* sc = &attn_s[tid * KNN];
    float mx = sc[0];
#pragma unroll
    for (int k = 1; k < KNN; ++k) mx = fmaxf(mx, sc[k]);
    float e_[KNN], ssum = 0.0f;
#pragma unroll
    for (int k = 0; k < KNN; ++k) { e_[k] = __expf(sc[k] - mx); ssum += e_[k]; }
    const float inv = 1.0f / ssum;
#pragma unroll
    for (int k = 0; k < KNN; ++k) sc[k] = e_[k] * inv;
  }
  __syncthreads();
  if (tid < TILE * KNN) {
    const int m = tid / KNN, k = tid % KNN;
    float w = 0.0f;
#pragma unroll
    for (int h = 0; h < HEADS; ++h) w += attn_s[(m * HEADS + h) * KNN + k];
    wavg[tid] = w * (1.0f / HEADS);
  }
  __syncthreads();

  // ---- Stage 4: vbar[m,h,:] = (sum_k attn*k_raw) - x   (softmax sums to 1) ----
  for (int t = tid; t < TILE * HEADS * EMB; t += 256) {
    const int m = t / (HEADS * EMB);
    const int h = (t / EMB) % HEADS;
    const int e = t & 255;
    const float* aw = &attn_s[(m * HEADS + h) * KNN];
    float acc = 0.0f;
#pragma unroll
    for (int k = 0; k < KNN; ++k)
      acc += aw[k] * __bfloat162float(k_bf[(m * KNN + k) * EMB + e]);
    acc -= __bfloat162float(x_bf[m * EMB + e]);
    big[t] = __float2bfloat16(acc);                              // overwrite qk
  }
  __syncthreads();

  // ---- Stage 5: out_attn[:, h*32 + dt*16 ..] = vbar_h[16,256] @ Wv_h^T ----
#pragma unroll
  for (int j = 0; j < 2; ++j) {
    const int job = wave * 2 + j;
    const int h = job >> 1, dt = job & 1;
    v8f acc = {};
#pragma unroll
    for (int kk = 0; kk < 8; ++kk) {
      v16bf a  = load_A(big + h * EMB, HEADS * EMB, kk * 32, lane);
      v16bf bm = load_B(Wv + (size_t)(h * 32 + dt * 16) * EMB, EMB, kk * 32, lane);
      acc = wmma_bf16(acc, a, bm);
    }
    const int n  = h * 32 + dt * 16 + (lane & 15);
    const int mb = (lane >> 4) * 8;
#pragma unroll
    for (int r = 0; r < 8; ++r)
      out_f[(mb + r) * EMB + n] = acc[r];
  }
  __syncthreads();
  for (int i = tid; i < TILE * EMB; i += 256)                    // reuse q_bf as out_bf
    q_bf[i] = __float2bfloat16(out_f[i]);
  __syncthreads();

  // ---- Stage 6: x_out = x + out_attn @ Wout^T ----
#pragma unroll
  for (int j = 0; j < 2; ++j) {
    const int nt = wave * 2 + j;
    v8f acc = {};
#pragma unroll
    for (int kk = 0; kk < 8; ++kk) {
      v16bf a  = load_A(q_bf, EMB, kk * 32, lane);
      v16bf bm = load_B(Wo + (size_t)nt * 16 * EMB, EMB, kk * 32, lane);
      acc = wmma_bf16(acc, a, bm);
    }
    const int n  = nt * 16 + (lane & 15);
    const int mb = (lane >> 4) * 8;
#pragma unroll
    for (int r = 0; r < 8; ++r) {
      const int m = mb + r;
      xout[((size_t)b * NPTS + v0 + m) * EMB + n] =
          xb[(size_t)(v0 + m) * EMB + n] + acc[r];
    }
  }

  // ---- Stage 7: scatter mean numerators/denominators ----
  if (tid < TILE * KNN) {
    const int j = idx_s[tid];
    atomicAdd(&sums[(size_t)b * NPTS + j], wavg[tid]);
    atomicAdd(&counts[(size_t)b * NPTS + j], 1.0f);
  }
}

__global__ __launch_bounds__(256)
void finalize_kernel(const float* __restrict__ sums, const float* __restrict__ counts,
                     float* __restrict__ xw) {
  const int i = blockIdx.x * 256 + threadIdx.x;
  if (i < BATCH * NPTS) xw[i] = sums[i] / (counts[i] + 1.0f);
}

// ---------------------------------------------------------------------------
extern "C" void kernel_launch(void* const* d_in, const int* in_sizes, int n_in,
                              void* d_out, int out_size, void* d_ws, size_t ws_size,
                              hipStream_t stream) {
  (void)in_sizes; (void)n_in; (void)out_size; (void)ws_size;
  const float* x        = (const float*)d_in[0];
  const float* xv       = (const float*)d_in[1];
  const float* in_proj  = (const float*)d_in[2];
  const float* out_proj = (const float*)d_in[3];
  float* xout = (float*)d_out;
  float* xw   = xout + (size_t)BATCH * NPTS * EMB;

  char* ws = (char*)d_ws;
  int* idx_ws = (int*)ws;                    ws += (size_t)BATCH * NPTS * KNN * sizeof(int);
  __hip_bfloat16* Wq  = (__hip_bfloat16*)ws; ws += (size_t)EMB * EMB * 2;
  __hip_bfloat16* WkT = (__hip_bfloat16*)ws; ws += (size_t)EMB * EMB * 2;
  __hip_bfloat16* Wv  = (__hip_bfloat16*)ws; ws += (size_t)EMB * EMB * 2;
  __hip_bfloat16* Wo  = (__hip_bfloat16*)ws; ws += (size_t)EMB * EMB * 2;
  float* sums   = (float*)ws;                ws += (size_t)BATCH * NPTS * sizeof(float);
  float* counts = (float*)ws;                // adjacent to sums -> one zero pass

  prep_weights<<<(EMB * EMB) / 256, 256, 0, stream>>>(in_proj, out_proj, Wq, WkT, Wv, Wo);
  zero_kernel<<<(2 * BATCH * NPTS + 255) / 256, 256, 0, stream>>>(sums, 2 * BATCH * NPTS);
  knn_kernel<<<BATCH * (NPTS / 256), 256, 0, stream>>>(xv, idx_ws);

  const size_t smem =
      (size_t)TILE * EMB * 2 +                 // x_bf
      (size_t)TILE * KNN * EMB * 2 +           // k_bf
      (size_t)TILE * EMB * 2 +                 // q_bf
      (size_t)TILE * HEADS * EMB * 2 +         // big
      (size_t)TILE * EMB * 4 +                 // out_f
      (size_t)TILE * HEADS * KNN * 4 +         // attn_s
      (size_t)TILE * KNN * 4 +                 // wavg
      (size_t)TILE * KNN * 4;                  // idx_s
  hipFuncSetAttribute((const void*)mha_kernel,
                      hipFuncAttributeMaxDynamicSharedMemorySize, (int)smem);
  mha_kernel<<<BATCH * NPTS / TILE, 256, smem, stream>>>(x, idx_ws, Wq, WkT, Wv, Wo,
                                                         xout, sums, counts);
  finalize_kernel<<<(BATCH * NPTS + 255) / 256, 256, 0, stream>>>(sums, counts, xw);
}
